// BatchedLSTM_23330262352282
// MI455X (gfx1250) — compile-verified, matrix operand order checked by reading, not run on hardware
//
#include <hip/hip_runtime.h>
#include <math.h>

typedef __bf16 v8bf  __attribute__((ext_vector_type(8)));
typedef __bf16 v16bf __attribute__((ext_vector_type(16)));
typedef float  v8f   __attribute__((ext_vector_type(8)));

#define Bdim 16
#define Fdim 2048
#define Cdim 512
#define Hdim 512
#define Kdim (Cdim + Hdim)

// LDS row pitch in 16-bit elements: 32 data + 8 pad = 80 bytes (16B aligned,
// 20-dword stride => conflict-free ds_read_b128 across the 16 fragment rows)
#define APITCH 40
#define BPITCH 40

// fp32 pair -> packed bf16 (RNE), low16 = a, high16 = b.
// Rounding in 32-bit VALU, merge with a single v_perm_b32.
__device__ __forceinline__ unsigned int pack2(float a, float b) {
    unsigned int ua = __float_as_uint(a);
    unsigned int ub = __float_as_uint(b);
    ua += 0x7FFFu + ((ua >> 16) & 1u);
    ub += 0x7FFFu + ((ub >> 16) & 1u);
    // perm operand order: ua -> bytes 0..3, ub -> bytes 4..7
    return __builtin_amdgcn_perm(ub, ua, 0x07060302u);
}
__device__ __forceinline__ float sigmoidf(float v) {
    return 1.0f / (1.0f + __expf(-v));
}

__global__ __launch_bounds__(256) void lstm_step_wmma_bf16(
    const float* __restrict__ x,       // (B, F, C)
    const float* __restrict__ hidden,  // (B, H, F)
    const float* __restrict__ cell,    // (B, H, F)
    const float* __restrict__ Wf, const float* __restrict__ bF,
    const float* __restrict__ Wi, const float* __restrict__ bI,
    const float* __restrict__ Wo, const float* __restrict__ bO,
    const float* __restrict__ Wg, const float* __restrict__ bG,
    float* __restrict__ out)           // [ (B,F,H) | (B,H,F) | (B,H,F) ]
{
    __shared__ alignas(16) unsigned short ldsA[256 * APITCH]; // 4 gates x 64 h-rows x 32 k (bf16)
    __shared__ alignas(16) unsigned short ldsB[64 * BPITCH];  // 64 f-rows x 32 k (bf16), f-major

    const int tid    = threadIdx.x;
    const int lane   = tid & 31;
    const int lane16 = lane & 15;
    const int hihalf = lane >> 4;      // wave32 half select (K+8 / M+8 per ISA layout)
    const int wave   = tid >> 5;
    const int msub   = wave >> 1;      // 16-row h subtile 0..3
    const int nhalf  = wave & 1;       // which 32-column f half

    const int fTile = blockIdx.x * 64;
    const int hTile = blockIdx.y * 64;
    const int b     = blockIdx.z;

    v8f acc[4][2];
#pragma unroll
    for (int g = 0; g < 4; ++g)
#pragma unroll
        for (int ns = 0; ns < 2; ++ns)
            acc[g][ns] = (v8f){0.f, 0.f, 0.f, 0.f, 0.f, 0.f, 0.f, 0.f};

    // Cooperative A-load assignment: one (gate, h-row) per thread.
    // Gate order matches reference concat [W_f, W_i, W_g, W_o].
    const int agate = tid >> 6;
    const int am    = tid & 63;
    const float* Wrow =
        (agate == 0 ? Wf : agate == 1 ? Wi : agate == 2 ? Wg : Wo)
        + (size_t)(hTile + am) * Kdim;

    for (int k0 = 0; k0 < Kdim; k0 += 32) {
        // ---- stage A tile: W rows, k-contiguous, fp32 -> bf16 ----
        {
            const float* src = Wrow + k0;
            unsigned short* dst = &ldsA[tid * APITCH];
            __builtin_prefetch(src + 32, 0, 1);   // global_prefetch_b8 next K chunk
#pragma unroll
            for (int j = 0; j < 8; ++j) {
                float4 v = *(const float4*)(src + 4 * j);
                uint2 p;
                p.x = pack2(v.x, v.y);
                p.y = pack2(v.z, v.w);
                *(uint2*)&dst[4 * j] = p;
            }
        }
        // ---- stage B tile (XH), stored f-major: ldsB[f][k] ----
        if (k0 < Cdim) {
            // x part: x[b, fTile+n, k0+kl] — k contiguous in memory
            const int n  = tid >> 2;
            const int kl = (tid & 3) * 8;
            const float* src = x + ((size_t)b * Fdim + (fTile + n)) * Cdim + k0 + kl;
            float4 v0 = *(const float4*)(src);
            float4 v1 = *(const float4*)(src + 4);
            unsigned short* dst = &ldsB[n * BPITCH + kl];
            uint2 p0; p0.x = pack2(v0.x, v0.y); p0.y = pack2(v0.z, v0.w);
            uint2 p1; p1.x = pack2(v1.x, v1.y); p1.y = pack2(v1.z, v1.w);
            *(uint2*)&dst[0] = p0;
            *(uint2*)&dst[4] = p1;
        } else {
            // hidden part: rows k=2*kp and k=2*kp+1 (f contiguous), pack
            // vertically -> 4 aligned ds_store_b32 into ldsB[f][k]
            const int kp = tid >> 4;          // 0..15 -> k pair
            const int n0 = (tid & 15) * 4;    // 4 f columns
            const float* src =
                hidden + ((size_t)b * Hdim + (k0 - Cdim + 2 * kp)) * Fdim + fTile + n0;
            float4 v0 = *(const float4*)(src);          // row k
            float4 v1 = *(const float4*)(src + Fdim);   // row k+1
            *(unsigned int*)&ldsB[(n0 + 0) * BPITCH + 2 * kp] = pack2(v0.x, v1.x);
            *(unsigned int*)&ldsB[(n0 + 1) * BPITCH + 2 * kp] = pack2(v0.y, v1.y);
            *(unsigned int*)&ldsB[(n0 + 2) * BPITCH + 2 * kp] = pack2(v0.z, v1.z);
            *(unsigned int*)&ldsB[(n0 + 3) * BPITCH + 2 * kp] = pack2(v0.w, v1.w);
        }
        __syncthreads();

        // ---- build fragments per documented 16-bit 16x32 layout:
        // lane holds row (lane&15); VGPR0-3 <- K = hihalf*8 + 0..7,
        // VGPR4-7 <- K = 16 + hihalf*8 + 0..7  => two ds_read_b128 per fragment.
        v16bf afrag[4];
#pragma unroll
        for (int g = 0; g < 4; ++g) {
            const unsigned short* ap =
                &ldsA[(g * 64 + msub * 16 + lane16) * APITCH + hihalf * 8];
            v8bf lo = *(const v8bf*)(ap);
            v8bf hi = *(const v8bf*)(ap + 16);
            afrag[g] = __builtin_shufflevector(lo, hi,
                0, 1, 2, 3, 4, 5, 6, 7, 8, 9, 10, 11, 12, 13, 14, 15);
        }
        v16bf bfrag[2];
#pragma unroll
        for (int ns = 0; ns < 2; ++ns) {
            const unsigned short* bp =
                &ldsB[(nhalf * 32 + ns * 16 + lane16) * BPITCH + hihalf * 8];
            v8bf lo = *(const v8bf*)(bp);
            v8bf hi = *(const v8bf*)(bp + 16);
            bfrag[ns] = __builtin_shufflevector(lo, hi,
                0, 1, 2, 3, 4, 5, 6, 7, 8, 9, 10, 11, 12, 13, 14, 15);
        }

        // ---- 8 WMMAs per wave per K-step ----
#pragma unroll
        for (int g = 0; g < 4; ++g)
#pragma unroll
            for (int ns = 0; ns < 2; ++ns)
                acc[g][ns] = __builtin_amdgcn_wmma_f32_16x16x32_bf16(
                    false, afrag[g], false, bfrag[ns],
                    (short)0, acc[g][ns], false, false);

        __syncthreads();
    }

    // ---- fused LSTM epilogue: all four gates for (h,f) live in this lane ----
    const size_t OFF1 = (size_t)Bdim * Fdim * Hdim;
#pragma unroll
    for (int ns = 0; ns < 2; ++ns) {
        const int f = fTile + nhalf * 32 + ns * 16 + lane16;
#pragma unroll
        for (int i = 0; i < 8; ++i) {
            // C/D layout: lanes 0-15 -> M=i, lanes 16-31 -> M=i+8
            const int h = hTile + msub * 16 + i + 8 * hihalf;
            const float gFv = acc[0][ns][i] + bF[h];
            const float gIv = acc[1][ns][i] + bI[h];
            const float gGv = acc[2][ns][i] + bG[h];
            const float gOv = acc[3][ns][i] + bO[h];
            const float ft = sigmoidf(gFv);
            const float it = sigmoidf(gIv);
            const float gt = tanhf(gGv);
            const float ot = sigmoidf(gOv);
            const size_t idx_hf = ((size_t)b * Hdim + h) * Fdim + f;
            const float c_new = cell[idx_hf] * ft + it * gt;
            const float h_new = tanhf(c_new) * ot;
            out[((size_t)b * Fdim + f) * Hdim + h] = h_new;  // (B,F,H)
            out[OFF1 + idx_hf]        = h_new;               // (B,H,F)
            out[2 * OFF1 + idx_hf]    = c_new;               // (B,H,F)
        }
    }
}

extern "C" void kernel_launch(void* const* d_in, const int* in_sizes, int n_in,
                              void* d_out, int out_size, void* d_ws, size_t ws_size,
                              hipStream_t stream) {
    (void)in_sizes; (void)n_in; (void)d_ws; (void)ws_size; (void)out_size;
    const float* x      = (const float*)d_in[0];
    const float* hidden = (const float*)d_in[1];
    const float* cell   = (const float*)d_in[2];
    const float* Wf     = (const float*)d_in[3];
    const float* bF     = (const float*)d_in[4];
    const float* Wi     = (const float*)d_in[5];
    const float* bI     = (const float*)d_in[6];
    const float* Wo     = (const float*)d_in[7];
    const float* bO     = (const float*)d_in[8];
    const float* Wg     = (const float*)d_in[9];
    const float* bG     = (const float*)d_in[10];

    dim3 grid(Fdim / 64, Hdim / 64, Bdim);
    lstm_step_wmma_bf16<<<grid, 256, 0, stream>>>(
        x, hidden, cell, Wf, bF, Wi, bI, Wo, bO, Wg, bG, (float*)d_out);
}
